// RubikSeq2SeqTransformer_30227979829284
// MI455X (gfx1250) — compile-verified, matrix operand order checked by reading, not run on hardware
//
#include <hip/hip_runtime.h>

#define DD   256
#define HH   8
#define LLAY 4
#define DFF  1024
#define BB   128
#define SS   50
#define TTS  49
#define NM   21
#define ME   (BB*SS)    // 6400 encoder tokens
#define MDX  (BB*TTS)   // 6272 decoder tokens
#define NT   4          // 16-wide N-subtiles per wave (A-fragment reuse)
#define MT   2          // 16-high M-subtiles per wave (B-fragment reuse)

typedef __attribute__((ext_vector_type(16))) __bf16 v16bf;
typedef __attribute__((ext_vector_type(8)))  float  v8f;

__device__ __forceinline__ unsigned short f2bf(float f) {
  unsigned u = __builtin_bit_cast(unsigned, f);
  u += 0x7FFFu + ((u >> 16) & 1u);            // round-to-nearest-even
  return (unsigned short)(u >> 16);
}

union Frag32 { uint4 u[2]; v16bf v; };

// ---------------------------------------------------------------------------
// fp32 -> bf16 bulk convert (weights, once per launch)
// ---------------------------------------------------------------------------
__global__ void cvt_bf16_kernel(const float* __restrict__ s, unsigned short* __restrict__ d, int n) {
  int i = blockIdx.x * 256 + threadIdx.x;
  if (i < n) d[i] = f2bf(s[i]);
}

// ---------------------------------------------------------------------------
// WMMA bf16 NT-GEMM:  C[M,N] = A[M,K] * Bw[N,K]^T (+bias) (+relu)
// One wave = 32x64 output strip (MT=2 x NT=4 16x16 tiles): per K-step,
// 2 A-frags + 4 B-frags feed 8 WMMAs (1.5 b128 loads per matrix op).
// All loads of a K-step are issued before the WMMA burst so the scheduler
// can keep multiple loads in flight (partial s_wait_loadcnt) instead of
// draining per-WMMA.
// A-frag per ISA 16-bit 16x32 layout: lane&15 = row, lanes<16 K{0..7,16..23},
// lanes>=16 K{8..15,24..31}.  B-frag mirrors with N in place of M.
// C/D layout: vgpr r, lanes 0-15 -> (M=r, N=lane), lanes 16-31 -> (M=8+r).
// ---------------------------------------------------------------------------
__global__ void __launch_bounds__(128)
gemm_bf16_nt(const unsigned short* __restrict__ A, const unsigned short* __restrict__ Bw,
             const float* __restrict__ bias, float* __restrict__ C,
             unsigned short* __restrict__ Cb, int N, int K, int relu)
{
  const int lane   = threadIdx.x;                      // 0..31
  const int Ntiles = (N + 15) >> 4;
  const int wtile  = blockIdx.y * blockDim.y + threadIdx.y;   // group-of-NT index
  if (wtile * NT >= Ntiles) return;                    // wave-uniform
  const int ntile0 = wtile * (NT * 16);
  const int mtile  = blockIdx.x * (MT * 16);

  const int r     = lane & 15;
  const int half  = lane >> 4;
  const int kbase = half * 8;

  size_t arow[MT];
#pragma unroll
  for (int mt = 0; mt < MT; ++mt)
    arow[mt] = (size_t)(mtile + mt * 16 + r) * (size_t)K;
  size_t brow[NT];
#pragma unroll
  for (int j = 0; j < NT; ++j) {
    int bn = ntile0 + j * 16 + r;
    if (bn > N - 1) bn = N - 1;                        // clamp for ragged N
    brow[j] = (size_t)bn * (size_t)K;
  }

  v8f acc[MT][NT] = {};
  for (int k0 = 0; k0 < K; k0 += 32) {
    Frag32 fa[MT], fb[NT];
#pragma unroll
    for (int mt = 0; mt < MT; ++mt) {
      fa[mt].u[0] = *(const uint4*)(A + arow[mt] + k0 + kbase);
      fa[mt].u[1] = *(const uint4*)(A + arow[mt] + k0 + kbase + 16);
    }
#pragma unroll
    for (int j = 0; j < NT; ++j) {
      fb[j].u[0] = *(const uint4*)(Bw + brow[j] + k0 + kbase);
      fb[j].u[1] = *(const uint4*)(Bw + brow[j] + k0 + kbase + 16);
    }
    __builtin_prefetch(A + arow[0] + k0 + 128, 0, 3);  // WGP-scope speculative prefetch
#pragma unroll
    for (int mt = 0; mt < MT; ++mt)
#pragma unroll
      for (int j = 0; j < NT; ++j)
        acc[mt][j] = __builtin_amdgcn_wmma_f32_16x16x32_bf16(false, fa[mt].v, false, fb[j].v,
                                                             (short)0, acc[mt][j], false, false);
  }

#pragma unroll
  for (int mt = 0; mt < MT; ++mt)
#pragma unroll
    for (int j = 0; j < NT; ++j) {
      const int ccol = ntile0 + j * 16 + r;
      if (ccol < N) {
        const float bv = bias ? bias[ccol] : 0.0f;
#pragma unroll
        for (int rr = 0; rr < 8; ++rr) {
          const int crow = mtile + mt * 16 + half * 8 + rr;
          float v = acc[mt][j][rr] + bv;
          if (relu) v = fmaxf(v, 0.0f);
          C[(size_t)crow * N + ccol] = v;
          if (Cb) Cb[(size_t)crow * N + ccol] = f2bf(v);
        }
      }
    }
}

// ---------------------------------------------------------------------------
// LayerNorm (optional residual add): Y = LN(X [+R]) * g + b ; optional bf16 copy
// one 256-thread block per row (D=256).  In-place X==Y safe.
// ---------------------------------------------------------------------------
__global__ void __launch_bounds__(256)
ln_kernel(const float* __restrict__ X, const float* __restrict__ R,
          const float* __restrict__ g, const float* __restrict__ bta,
          float* __restrict__ Y, unsigned short* __restrict__ Yb)
{
  const int row = blockIdx.x, d = threadIdx.x;
  float v = X[(size_t)row * DD + d];
  if (R) v += R[(size_t)row * DD + d];
  __shared__ float red[DD];
  red[d] = v; __syncthreads();
  for (int s2 = 128; s2 > 0; s2 >>= 1) { if (d < s2) red[d] += red[d + s2]; __syncthreads(); }
  const float mean = red[0] * (1.0f / DD);
  __syncthreads();
  const float dv = v - mean;
  red[d] = dv * dv; __syncthreads();
  for (int s2 = 128; s2 > 0; s2 >>= 1) { if (d < s2) red[d] += red[d + s2]; __syncthreads(); }
  const float var = red[0] * (1.0f / DD);
  const float y = dv * rsqrtf(var + 1e-5f) * g[d] + bta[d];
  if (Y)  Y[(size_t)row * DD + d]  = y;
  if (Yb) Yb[(size_t)row * DD + d] = f2bf(y);
}

// ---------------------------------------------------------------------------
// Face encoder: lin(9->D) + 3x3 SAME conv on 3x3 faces + relu + spatial mean,
// averaged over 6 faces, + positional embedding.  One block per (b,s).
// ---------------------------------------------------------------------------
__global__ void __launch_bounds__(256)
face_kernel(const float* __restrict__ src, const float* __restrict__ Wf,
            const float* __restrict__ bf,  const float* __restrict__ Wc,
            const float* __restrict__ bc,  const float* __restrict__ spos,
            float* __restrict__ X, unsigned short* __restrict__ Xb)
{
  const int bs = blockIdx.x;                // b*S + s
  const int b = bs / SS, s = bs % SS;
  const int d = threadIdx.x;
  __shared__ float cube[54];
  if (d < 54) cube[d] = src[(size_t)bs * 55 + d];
  __syncthreads();

  float wf[9], wc[9];
#pragma unroll
  for (int i = 0; i < 9; ++i) { wf[i] = Wf[d * 9 + i]; wc[i] = Wc[d * 9 + i]; }
  const float bfv = bf[d], bcv = bc[d];

  float acc = 0.f;
  for (int f = 0; f < 6; ++f) {
    const float* cu = cube + f * 9;
    float lin = bfv;
#pragma unroll
    for (int i = 0; i < 9; ++i) lin += cu[i] * wf[i];
    float cf = 0.f;
#pragma unroll
    for (int y = 0; y < 3; ++y)
#pragma unroll
      for (int x = 0; x < 3; ++x) {
        float c = bcv;
#pragma unroll
        for (int ky = 0; ky < 3; ++ky)
#pragma unroll
          for (int kx = 0; kx < 3; ++kx) {
            const int iy = y + ky - 1, ix = x + kx - 1;
            if (iy >= 0 && iy < 3 && ix >= 0 && ix < 3)
              c += cu[iy * 3 + ix] * wc[ky * 3 + kx];
          }
        cf += fmaxf(c, 0.f);
      }
    acc += 0.5f * (lin + cf * (1.f / 9.f));
  }
  acc *= (1.f / 6.f);
  const float v = acc + spos[s * DD + d];
  const size_t tok = (size_t)(s * BB + b);  // seq-first token index
  X[tok * DD + d]  = v;
  Xb[tok * DD + d] = f2bf(v);
}

// ---------------------------------------------------------------------------
// Decoder token embedding + positional
// ---------------------------------------------------------------------------
__global__ void __launch_bounds__(256)
embed_kernel(const int* __restrict__ tgt, const float* __restrict__ emb,
             const float* __restrict__ tpos, float* __restrict__ Y,
             unsigned short* __restrict__ Yb)
{
  const int tok = blockIdx.x;               // t*B + b
  const int t = tok / BB, b = tok % BB;
  const int d = threadIdx.x;
  const int idx = tgt[b * TTS + t];
  const float v = emb[idx * DD + d] + tpos[t * DD + d];
  Y[(size_t)tok * DD + d]  = v;
  Yb[(size_t)tok * DD + d] = f2bf(v);
}

// ---------------------------------------------------------------------------
// Flash-style attention: one wave per (query-row, batch, head); lane = head dim.
// Online softmax, 32-lane xor-shuffle reduction.  Writes bf16 (GEMM input).
// ---------------------------------------------------------------------------
__global__ void __launch_bounds__(256)
attn_kernel(const float* __restrict__ Q, int qstride,
            const float* __restrict__ Kp, const float* __restrict__ Vp, int kvstride,
            unsigned short* __restrict__ Ob, int Tq, int Tk, int causal)
{
  const int widx = threadIdx.x >> 5, lane = threadIdx.x & 31;
  const int gid = blockIdx.x * 8 + widx;
  if (gid >= Tq * BB * HH) return;
  const int h = gid % HH;
  const int rem = gid / HH;
  const int b = rem % BB;
  const int s = rem / BB;

  const float q = Q[(size_t)(s * BB + b) * qstride + h * 32 + lane] * 0.17677669529663687f;
  float m = -1e30f, l = 0.f, o = 0.f;
  const int tend = causal ? (s + 1) : Tk;
  for (int t = 0; t < tend; ++t) {
    const size_t kr = (size_t)(t * BB + b) * kvstride + h * 32 + lane;
    float p = q * Kp[kr];
#pragma unroll
    for (int mm = 16; mm >= 1; mm >>= 1) p += __shfl_xor(p, mm, 32);
    const float mn   = fmaxf(m, p);
    const float corr = __expf(m - mn);
    const float e    = __expf(p - mn);
    l = l * corr + e;
    o = o * corr + e * Vp[kr];
    m = mn;
  }
  Ob[(size_t)(s * BB + b) * DD + h * 32 + lane] = f2bf(o / l);
}

// ---------------------------------------------------------------------------
// (T,B,NM) -> (B,T,NM) logits remap
// ---------------------------------------------------------------------------
__global__ void out_transpose(const float* __restrict__ tmp, float* __restrict__ out)
{
  const int i = blockIdx.x * 256 + threadIdx.x;
  if (i >= MDX * NM) return;
  const int n = i % NM;
  const int tok = i / NM;                   // t*B + b
  const int t = tok / BB, b = tok % BB;
  out[((size_t)b * TTS + t) * NM + n] = tmp[i];
}

// ---------------------------------------------------------------------------
// host orchestration
// ---------------------------------------------------------------------------
enum {
  IN_SRC = 0, IN_TGT, IN_WFACE, IN_BFACE, IN_WCONV, IN_BCONV, IN_SRCPOS, IN_TGTPOS,
  IN_TGTEMB, IN_WOUT, IN_BOUT, IN_ENC_NG, IN_ENC_NB, IN_DEC_NG, IN_DEC_NB,
  IN_ESA_WQKV, IN_ESA_BQKV, IN_ESA_WO, IN_ESA_BO,
  IN_E_W1, IN_E_B1, IN_E_W2, IN_E_B2,
  IN_E_LN1G, IN_E_LN1B, IN_E_LN2G, IN_E_LN2B,
  IN_DSA_WQKV, IN_DSA_BQKV, IN_DSA_WO, IN_DSA_BO,
  IN_DCA_WQKV, IN_DCA_BQKV, IN_DCA_WO, IN_DCA_BO,
  IN_D_W1, IN_D_B1, IN_D_W2, IN_D_B2,
  IN_D_LN1G, IN_D_LN1B, IN_D_LN2G, IN_D_LN2B, IN_D_LN3G, IN_D_LN3B
};

extern "C" void kernel_launch(void* const* d_in, const int* in_sizes, int n_in,
                              void* d_out, int out_size, void* d_ws, size_t ws_size,
                              hipStream_t stream)
{
  (void)in_sizes; (void)n_in; (void)out_size; (void)ws_size;
  #define FP(i) ((const float*)d_in[i])

  char* base = (char*)d_ws;
  size_t off = 0;
  auto alloc = [&](size_t bytes) -> void* {
    void* p = base + off;
    off += (bytes + 255) & ~(size_t)255;
    return p;
  };

  // bf16 weight shadows
  unsigned short* wEQkv = (unsigned short*)alloc((size_t)LLAY*3*DD*DD*2);
  unsigned short* wEWo  = (unsigned short*)alloc((size_t)LLAY*DD*DD*2);
  unsigned short* wEW1  = (unsigned short*)alloc((size_t)LLAY*DFF*DD*2);
  unsigned short* wEW2  = (unsigned short*)alloc((size_t)LLAY*DD*DFF*2);
  unsigned short* wDQkv = (unsigned short*)alloc((size_t)LLAY*3*DD*DD*2);
  unsigned short* wDWo  = (unsigned short*)alloc((size_t)LLAY*DD*DD*2);
  unsigned short* wCQkv = (unsigned short*)alloc((size_t)LLAY*3*DD*DD*2);
  unsigned short* wCWo  = (unsigned short*)alloc((size_t)LLAY*DD*DD*2);
  unsigned short* wDW1  = (unsigned short*)alloc((size_t)LLAY*DFF*DD*2);
  unsigned short* wDW2  = (unsigned short*)alloc((size_t)LLAY*DD*DFF*2);
  unsigned short* wOut  = (unsigned short*)alloc((size_t)NM*DD*2);
  // activations
  float*          X    = (float*)alloc((size_t)ME*DD*4);
  unsigned short* XB   = (unsigned short*)alloc((size_t)ME*DD*2);
  float*          SCR  = (float*)alloc((size_t)ME*DFF*4);           // qkv / ffn hidden
  unsigned short* SCRB = (unsigned short*)alloc((size_t)ME*DFF*2);  // hidden/attn-out bf16
  float*          P    = (float*)alloc((size_t)ME*DD*4);            // projection
  float*          QC   = (float*)alloc((size_t)MDX*DD*4);           // cross-attn Q
  float*          KVM  = (float*)alloc((size_t)ME*2*DD*4);          // cross-attn K,V
  unsigned short* MEMB = (unsigned short*)alloc((size_t)ME*DD*2);   // encoder memory bf16
  float*          OT   = (float*)alloc((size_t)MDX*NM*4);           // pre-transpose logits

  auto cvt = [&](const float* s, unsigned short* d, size_t n) {
    cvt_bf16_kernel<<<dim3((unsigned)((n + 255) / 256)), dim3(256), 0, stream>>>(s, d, (int)n);
  };
  cvt(FP(IN_ESA_WQKV), wEQkv, (size_t)LLAY*3*DD*DD);
  cvt(FP(IN_ESA_WO),   wEWo,  (size_t)LLAY*DD*DD);
  cvt(FP(IN_E_W1),     wEW1,  (size_t)LLAY*DFF*DD);
  cvt(FP(IN_E_W2),     wEW2,  (size_t)LLAY*DD*DFF);
  cvt(FP(IN_DSA_WQKV), wDQkv, (size_t)LLAY*3*DD*DD);
  cvt(FP(IN_DSA_WO),   wDWo,  (size_t)LLAY*DD*DD);
  cvt(FP(IN_DCA_WQKV), wCQkv, (size_t)LLAY*3*DD*DD);
  cvt(FP(IN_DCA_WO),   wCWo,  (size_t)LLAY*DD*DD);
  cvt(FP(IN_D_W1),     wDW1,  (size_t)LLAY*DFF*DD);
  cvt(FP(IN_D_W2),     wDW2,  (size_t)LLAY*DD*DFF);
  cvt(FP(IN_WOUT),     wOut,  (size_t)NM*DD);

  auto gemm = [&](const unsigned short* A, const unsigned short* Bw, const float* bias,
                  float* C, unsigned short* Cb, int M, int N, int K, int relu) {
    const int ntgroups = (((N + 15) / 16) + NT - 1) / NT;   // 32x64 wave strips
    dim3 grid((unsigned)(M / (MT * 16)), (unsigned)((ntgroups + 3) / 4));
    dim3 block(32, 4);
    gemm_bf16_nt<<<grid, block, 0, stream>>>(A, Bw, bias, C, Cb, N, K, relu);
  };
  auto ln = [&](const float* Xi, const float* R, const float* g, const float* bta,
                float* Y, unsigned short* Yb, int rows) {
    ln_kernel<<<dim3((unsigned)rows), dim3(256), 0, stream>>>(Xi, R, g, bta, Y, Yb);
  };

  // --- face encoder -> X (S,B,D) seq-first + bf16 shadow ---
  face_kernel<<<dim3(BB*SS), dim3(256), 0, stream>>>(
      FP(IN_SRC), FP(IN_WFACE), FP(IN_BFACE), FP(IN_WCONV), FP(IN_BCONV),
      FP(IN_SRCPOS), X, XB);

  // --- encoder stack ---
  for (int i = 0; i < LLAY; ++i) {
    gemm(XB, wEQkv + (size_t)i*3*DD*DD, FP(IN_ESA_BQKV) + i*3*DD, SCR, nullptr, ME, 3*DD, DD, 0);
    attn_kernel<<<dim3(SS*BB*HH/8), dim3(256), 0, stream>>>(
        SCR, 3*DD, SCR + DD, SCR + 2*DD, 3*DD, SCRB, SS, SS, 0);
    gemm(SCRB, wEWo + (size_t)i*DD*DD, FP(IN_ESA_BO) + i*DD, P, nullptr, ME, DD, DD, 0);
    ln(X, P, FP(IN_E_LN1G) + i*DD, FP(IN_E_LN1B) + i*DD, X, XB, ME);
    gemm(XB, wEW1 + (size_t)i*DFF*DD, FP(IN_E_B1) + i*DFF, SCR, SCRB, ME, DFF, DD, 1);
    gemm(SCRB, wEW2 + (size_t)i*DD*DFF, FP(IN_E_B2) + i*DD, P, nullptr, ME, DD, DFF, 0);
    ln(X, P, FP(IN_E_LN2G) + i*DD, FP(IN_E_LN2B) + i*DD, X, XB, ME);
  }
  // memory = LN(x) -> bf16 only
  ln(X, nullptr, FP(IN_ENC_NG), FP(IN_ENC_NB), nullptr, MEMB, ME);

  // --- decoder embedding (X/XB reused for y) ---
  embed_kernel<<<dim3(MDX), dim3(256), 0, stream>>>(
      (const int*)d_in[IN_TGT], FP(IN_TGTEMB), FP(IN_TGTPOS), X, XB);

  // --- decoder stack ---
  for (int i = 0; i < LLAY; ++i) {
    // causal self-attention
    gemm(XB, wDQkv + (size_t)i*3*DD*DD, FP(IN_DSA_BQKV) + i*3*DD, SCR, nullptr, MDX, 3*DD, DD, 0);
    attn_kernel<<<dim3(TTS*BB*HH/8), dim3(256), 0, stream>>>(
        SCR, 3*DD, SCR + DD, SCR + 2*DD, 3*DD, SCRB, TTS, TTS, 1);
    gemm(SCRB, wDWo + (size_t)i*DD*DD, FP(IN_DSA_BO) + i*DD, P, nullptr, MDX, DD, DD, 0);
    ln(X, P, FP(IN_D_LN1G) + i*DD, FP(IN_D_LN1B) + i*DD, X, XB, MDX);
    // cross-attention: Q from y, K/V from memory (Wqkv row slices)
    gemm(XB, wCQkv + (size_t)i*3*DD*DD, FP(IN_DCA_BQKV) + i*3*DD, QC, nullptr, MDX, DD, DD, 0);
    gemm(MEMB, wCQkv + (size_t)i*3*DD*DD + (size_t)DD*DD, FP(IN_DCA_BQKV) + i*3*DD + DD,
         KVM, nullptr, ME, 2*DD, DD, 0);
    attn_kernel<<<dim3(TTS*BB*HH/8), dim3(256), 0, stream>>>(
        QC, DD, KVM, KVM + DD, 2*DD, SCRB, TTS, SS, 0);
    gemm(SCRB, wCWo + (size_t)i*DD*DD, FP(IN_DCA_BO) + i*DD, P, nullptr, MDX, DD, DD, 0);
    ln(X, P, FP(IN_D_LN2G) + i*DD, FP(IN_D_LN2B) + i*DD, X, XB, MDX);
    // FFN
    gemm(XB, wDW1 + (size_t)i*DFF*DD, FP(IN_D_B1) + i*DFF, SCR, SCRB, MDX, DFF, DD, 1);
    gemm(SCRB, wDW2 + (size_t)i*DD*DFF, FP(IN_D_B2) + i*DD, P, nullptr, MDX, DD, DFF, 0);
    ln(X, P, FP(IN_D_LN3G) + i*DD, FP(IN_D_LN3B) + i*DD, X, XB, MDX);
  }
  // final norm -> bf16 only, then output projection (N=21) and (T,B)->(B,T) remap
  ln(X, nullptr, FP(IN_DEC_NG), FP(IN_DEC_NB), nullptr, XB, MDX);
  gemm(XB, wOut, FP(IN_BOUT), OT, nullptr, MDX, NM, DD, 0);
  out_transpose<<<dim3((MDX*NM + 255) / 256), dim3(256), 0, stream>>>(OT, (float*)d_out);
  #undef FP
}
